// SCNN_36283883716813
// MI455X (gfx1250) — compile-verified, matrix operand order checked by reading, not run on hardware
//
#include <hip/hip_runtime.h>
#include <stdint.h>

typedef __bf16 bf16;
typedef __attribute__((ext_vector_type(16))) __bf16 v16bf;
typedef __attribute__((ext_vector_type(8)))  __bf16 v8bf;
typedef __attribute__((ext_vector_type(8)))  float  v8f;
typedef __attribute__((ext_vector_type(4)))  int    v4i;

#define BB   16
#define HH   36
#define WW   100
#define C0   512
#define C1   1024
#define C2   128
#define MM   (BB*HH*WW)      /* 57600 */
#define K1   (9*C0)          /* 4608  */
#define KMP  (9*C2)          /* 1152  */
#define NSEG 5
#define SEGOUT ((size_t)BB*288*800*NSEG)
#define LDA  72              /* padded LDS row stride (bf16 elems, 144B) */

// ----------------------------------------------------------- async LDS copy
#if defined(__has_builtin)
#if __has_builtin(__builtin_amdgcn_global_load_async_to_lds_b128) && \
    __has_builtin(__builtin_amdgcn_s_wait_asynccnt)
#define HAVE_ASYNC 1
#endif
#endif
#ifndef HAVE_ASYNC
#define HAVE_ASYNC 0
#endif

__device__ __forceinline__ void ld16_to_lds(bf16* ldst, const bf16* gsrc) {
#if HAVE_ASYNC
  __builtin_amdgcn_global_load_async_to_lds_b128(
      (__attribute__((address_space(1))) v4i*)gsrc,
      (__attribute__((address_space(3))) v4i*)ldst, 0, 0);
#else
  *(v4i*)ldst = *(const v4i*)gsrc;
#endif
}
__device__ __forceinline__ void wait_async_lds() {
#if HAVE_ASYNC
  __builtin_amdgcn_s_wait_asynccnt(0);
#endif
}
__device__ __forceinline__ void st16_zero(bf16* ldst) {
  v4i z = {0, 0, 0, 0};
  *(v4i*)ldst = z;
}
__device__ __forceinline__ void cvt8_to_lds(bf16* ldst, const float* src) {
  v8bf t;
  #pragma unroll
  for (int i = 0; i < 8; ++i) t[i] = (bf16)src[i];
  *(v8bf*)ldst = t;
}

// ---------------------------------------------------------------- conversions
__global__ void k_cvt_bf16(const float* __restrict__ s, bf16* __restrict__ d, size_t n) {
  size_t i  = (size_t)blockIdx.x*blockDim.x + threadIdx.x;
  size_t st = (size_t)gridDim.x*blockDim.x;
  for (; i < n; i += st) d[i] = (bf16)s[i];
}

// dst[n*K + k] = src[k*N + n]   (weights -> N-major bf16 for WMMA B staging)
__global__ void k_cvtT_bf16(const float* __restrict__ s, bf16* __restrict__ d, int K, int N) {
  size_t tot = (size_t)K*N;
  size_t i  = (size_t)blockIdx.x*blockDim.x + threadIdx.x;
  size_t st = (size_t)gridDim.x*blockDim.x;
  for (; i < tot; i += st) {
    int n = (int)(i / K), k = (int)(i % K);
    d[i] = (bf16)s[(size_t)k*N + n];
  }
}

__global__ void k_zero(float* p, size_t n) {
  size_t i  = (size_t)blockIdx.x*blockDim.x + threadIdx.x;
  size_t st = (size_t)gridDim.x*blockDim.x;
  for (; i < n; i += st) p[i] = 0.f;
}

// ---------------------------------------------------------------- WMMA GEMM
// C[M x N] = A[M x K] * B[K x N]; B given transposed (N-major: BT[n*K + k]).
// Block = 64 threads (2 waves); tile 64x64; BK=64, double-buffered LDS with
// async global->LDS staging overlapped with WMMA compute.
// Each wave computes a 32x64 strip = 8 accumulator tiles.
// MODE 0: plain bf16 A (lda == K), store C f32
// MODE 1: implicit-GEMM gather for conv1 (dilation-4 3x3 over bf16 features)
// MODE 2: message-passing step: gather prev row (f32 -> bf16),
//         epilogue: cur += relu(acc)   (in-place scan update)
struct GArgs {
  const bf16*  A;      // mode 0
  const bf16*  feat;   // mode 1 (B,H,W,C0 bf16)
  const float* prev;   // mode 2 base of previous row
  float*       cur;    // mode 2 base of current row
  int64_t sB, sP;      // mode 2 strides (elements): batch, position
  int L;               // mode 2 row length
  const bf16* BT;      // N x K
  float* C;            // mode 0/1 output (ldc == N)
  int M, N, K;
};

template<int MODE>
__global__ __launch_bounds__(64) void k_gemm(GArgs g) {
  __shared__ __align__(16) bf16 sA[2][64*LDA];
  __shared__ __align__(16) bf16 sB[2][64*LDA];
  const int tid  = threadIdx.x;     // 0..63
  const int lane = tid & 31;
  const int wave = tid >> 5;        // 0..1
  const int bm = blockIdx.x * 64;
  const int bn = blockIdx.y * 64;

  v8f acc[2][4];
  #pragma unroll
  for (int a = 0; a < 2; ++a)
    #pragma unroll
    for (int t = 0; t < 4; ++t) acc[a][t] = 0;

  // stage one BK=64 slice: thread `tid` fills A row `tid` and B column `tid`.
  auto stage = [&](int kk, int buf) {
    { // B column (always in range; contiguous 128B)
      const bf16* src = g.BT + (size_t)(bn + tid)*g.K + kk;
      bf16* dst = &sB[buf][tid*LDA];
      #pragma unroll
      for (int c = 0; c < 8; ++c) ld16_to_lds(dst + c*8, src + c*8);
    }
    bf16* dst = &sA[buf][tid*LDA];
    if (MODE == 0) {
      const bf16* src = g.A + (size_t)(bm + tid)*g.K + kk;
      #pragma unroll
      for (int c = 0; c < 8; ++c) ld16_to_lds(dst + c*8, src + c*8);
    } else if (MODE == 1) {
      const int gm = bm + tid;
      int b  = gm / (HH*WW);
      int hw = gm % (HH*WW);
      int h = hw / WW, w = hw % WW;
      int t9 = kk >> 9;                 // 512 ci per (kh,kw) tap; BK=64 divides 512
      int kh = t9 / 3, kw = t9 % 3;
      int ci = kk & (C0-1);
      int hh = h + 4*(kh-1);
      int ww = w + 4*(kw-1);
      if (hh >= 0 && hh < HH && ww >= 0 && ww < WW) {
        const bf16* src = g.feat + (((size_t)b*HH + hh)*WW + ww)*C0 + ci;
        #pragma unroll
        for (int c = 0; c < 8; ++c) ld16_to_lds(dst + c*8, src + c*8);
      } else {
        #pragma unroll
        for (int c = 0; c < 8; ++c) st16_zero(dst + c*8);
      }
    } else { // MODE 2: prev row is f32; convert while staging (no async)
      const int gm = bm + tid;
      int b = gm / g.L, pos = gm % g.L;
      #pragma unroll
      for (int c = 0; c < 8; ++c) {
        int kg  = kk + c*8;
        int tap = kg >> 7;              // 128 ci per tap
        int ci  = kg & (C2-1);
        int pp  = pos + tap - 4;        // SAME padding, width-9 kernel
        if (pp >= 0 && pp < g.L)
          cvt8_to_lds(dst + c*8, g.prev + (int64_t)b*g.sB + (int64_t)pp*g.sP + ci);
        else
          st16_zero(dst + c*8);
      }
    }
  };

  const int nstage = g.K >> 6;
  stage(0, 0);
  wait_async_lds();
  __syncthreads();

  for (int s = 0; s < nstage; ++s) {
    const int cur = s & 1;
    if (s + 1 < nstage) stage((s + 1) << 6, cur ^ 1);  // prefetch next slice

    const bf16* Ab = sA[cur];
    const bf16* Bb = sB[cur];
    const int kb = (lane >> 4) * 8;     // K-half select per ISA 16-bit layout
    #pragma unroll
    for (int ks = 0; ks < 64; ks += 32) {
      v16bf af[2];
      #pragma unroll
      for (int a = 0; a < 2; ++a) {
        const int row = wave*32 + a*16 + (lane & 15);
        v8bf lo = *(const v8bf*)(Ab + row*LDA + ks + kb);
        v8bf hi = *(const v8bf*)(Ab + row*LDA + ks + 16 + kb);
        af[a] = __builtin_shufflevector(lo, hi, 0,1,2,3,4,5,6,7,8,9,10,11,12,13,14,15);
      }
      #pragma unroll
      for (int t = 0; t < 4; ++t) {
        const int col = t*16 + (lane & 15);
        v8bf lo = *(const v8bf*)(Bb + col*LDA + ks + kb);
        v8bf hi = *(const v8bf*)(Bb + col*LDA + ks + 16 + kb);
        v16bf bfg = __builtin_shufflevector(lo, hi, 0,1,2,3,4,5,6,7,8,9,10,11,12,13,14,15);
        acc[0][t] = __builtin_amdgcn_wmma_f32_16x16x32_bf16(
            false, af[0], false, bfg, (short)0, acc[0][t], false, false);
        acc[1][t] = __builtin_amdgcn_wmma_f32_16x16x32_bf16(
            false, af[1], false, bfg, (short)0, acc[1][t], false, false);
      }
    }
    wait_async_lds();
    __syncthreads();
  }

  // ---- epilogue (D layout: VGPR r -> M = r + 8*(lane>=16), N = lane&15) ----
  #pragma unroll
  for (int a = 0; a < 2; ++a)
    #pragma unroll
    for (int t = 0; t < 4; ++t) {
      const int n = bn + t*16 + (lane & 15);
      #pragma unroll
      for (int r = 0; r < 8; ++r) {
        const int m = bm + wave*32 + a*16 + r + 8*(lane >> 4);
        if (MODE == 2) {
          int b = m / g.L, pos = m % g.L;
          float* dst = g.cur + (int64_t)b*g.sB + (int64_t)pos*g.sP + n;
          *dst = *dst + fmaxf(acc[a][t][r], 0.f);
        } else {
          g.C[(size_t)m*g.N + n] = acc[a][t][r];
        }
      }
    }
}

// ---------------------------------------------------------------- batch norm
__global__ void k_bn_reduce(const float* __restrict__ x, int M, int C, int rows,
                            float* __restrict__ sum, float* __restrict__ sumsq) {
  int row0 = blockIdx.x * rows;
  for (int c = threadIdx.x; c < C; c += blockDim.x) {
    float s = 0.f, q = 0.f;
    for (int r = 0; r < rows; ++r) {
      int m = row0 + r;
      if (m < M) { float v = x[(size_t)m*C + c]; s += v; q += v*v; }
    }
    atomicAdd(&sum[c], s);
    atomicAdd(&sumsq[c], q);
  }
}

template<int BFOUT>
__global__ void k_bn_apply(const float* __restrict__ x, const float* __restrict__ sum,
                           const float* __restrict__ sumsq, const float* __restrict__ scale,
                           const float* __restrict__ bias, float invN, int C, size_t total,
                           float* __restrict__ outf, bf16* __restrict__ outb) {
  size_t i  = (size_t)blockIdx.x*blockDim.x + threadIdx.x;
  size_t st = (size_t)gridDim.x*blockDim.x;
  for (; i < total; i += st) {
    int c = (int)(i % C);
    float m = sum[c]*invN;
    float v = sumsq[c]*invN - m*m;
    float y = (x[i]-m)*rsqrtf(v + 1e-5f)*scale[c] + bias[c];
    y = fmaxf(y, 0.f);
    if (BFOUT) outb[i] = (bf16)y; else outf[i] = y;
  }
}

// ---------------------------------------------------------------- seg head
__global__ void k_seg(const float* __restrict__ x, const float* __restrict__ w,
                      const float* __restrict__ b, float* __restrict__ seg) {
  int m = blockIdx.x*blockDim.x + threadIdx.x;
  if (m >= MM) return;
  float acc[NSEG];
  #pragma unroll
  for (int j = 0; j < NSEG; ++j) acc[j] = b[j];
  const float* xr = x + (size_t)m*C2;
  for (int c = 0; c < C2; ++c) {
    float v = xr[c];
    #pragma unroll
    for (int j = 0; j < NSEG; ++j) acc[j] += v*w[c*NSEG + j];
  }
  #pragma unroll
  for (int j = 0; j < NSEG; ++j) seg[(size_t)m*NSEG + j] = acc[j];
}

__global__ void k_resize(const float* __restrict__ seg, float* __restrict__ out) {
  const size_t total = (size_t)BB*288*800;
  size_t i = (size_t)blockIdx.x*blockDim.x + threadIdx.x;
  if (i >= total) return;
  int ow = (int)(i % 800);
  int tt = (int)(i / 800);
  int oh = tt % 288;
  int b  = tt / 288;
  float sh = (oh + 0.5f)*(36.f/288.f) - 0.5f;
  float sw = (ow + 0.5f)*(100.f/800.f) - 0.5f;
  sh = fminf(fmaxf(sh, 0.f), (float)(HH-1));
  sw = fminf(fmaxf(sw, 0.f), (float)(WW-1));
  int h0 = (int)sh, w0 = (int)sw;
  int h1 = (h0+1 < HH) ? h0+1 : HH-1;
  int w1 = (w0+1 < WW) ? w0+1 : WW-1;
  float fh = sh - (float)h0, fw = sw - (float)w0;
  const float* p00 = seg + (((size_t)b*HH + h0)*WW + w0)*NSEG;
  const float* p01 = seg + (((size_t)b*HH + h0)*WW + w1)*NSEG;
  const float* p10 = seg + (((size_t)b*HH + h1)*WW + w0)*NSEG;
  const float* p11 = seg + (((size_t)b*HH + h1)*WW + w1)*NSEG;
  #pragma unroll
  for (int j = 0; j < NSEG; ++j) {
    float v = (1.f-fh)*((1.f-fw)*p00[j] + fw*p01[j])
            +      fh *((1.f-fw)*p10[j] + fw*p11[j]);
    out[i*NSEG + j] = v;
  }
}

__global__ void k_softpool(const float* __restrict__ seg, float* __restrict__ p) {
  const int total = BB*18*50;
  int i = blockIdx.x*blockDim.x + threadIdx.x;
  if (i >= total) return;
  int w2 = i % 50;
  int t  = i / 50;
  int h2 = t % 18;
  int b  = t / 18;
  float acc[NSEG] = {0.f, 0.f, 0.f, 0.f, 0.f};
  for (int dh = 0; dh < 2; ++dh)
    for (int dw = 0; dw < 2; ++dw) {
      int h = 2*h2 + dh, w = 2*w2 + dw;
      const float* s = seg + (((size_t)b*HH + h)*WW + w)*NSEG;
      float mx = s[0];
      #pragma unroll
      for (int j = 1; j < NSEG; ++j) mx = fmaxf(mx, s[j]);
      float e[NSEG], ssum = 0.f;
      #pragma unroll
      for (int j = 0; j < NSEG; ++j) { e[j] = __expf(s[j]-mx); ssum += e[j]; }
      float inv = 1.f/ssum;
      #pragma unroll
      for (int j = 0; j < NSEG; ++j) acc[j] += e[j]*inv;
    }
  float* dst = p + (size_t)b*4500 + ((size_t)h2*50 + w2)*NSEG;
  #pragma unroll
  for (int j = 0; j < NSEG; ++j) dst[j] = acc[j]*0.25f;
}

__global__ void k_fc1(const float* __restrict__ p, const float* __restrict__ w,
                      const float* __restrict__ b, float* __restrict__ z) {
  int bb = blockIdx.x, o = threadIdx.x;   // 16 blocks x 128 threads
  float acc = b[o];
  const float* pr = p + (size_t)bb*4500;
  for (int k = 0; k < 4500; ++k) acc += pr[k]*w[(size_t)k*C2 + o];
  z[bb*C2 + o] = fmaxf(acc, 0.f);
}

__global__ void k_fc2(const float* __restrict__ z, const float* __restrict__ w,
                      const float* __restrict__ b, float* __restrict__ out) {
  int t = threadIdx.x;                    // 64 threads = 16 batches x 4 outputs
  if (t >= 64) return;
  int bb = t >> 2, j = t & 3;
  float acc = b[j];
  for (int o = 0; o < C2; ++o) acc += z[bb*C2 + o]*w[o*4 + j];
  out[t] = acc;
}

// ---------------------------------------------------------------- launch
extern "C" void kernel_launch(void* const* d_in, const int* in_sizes, int n_in,
                              void* d_out, int out_size, void* d_ws, size_t ws_size,
                              hipStream_t stream) {
  (void)in_sizes; (void)n_in; (void)out_size; (void)ws_size;
  const float* features = (const float*)d_in[0];
  const float* w_conv1  = (const float*)d_in[1];
  const float* bn1_s    = (const float*)d_in[2];
  const float* bn1_b    = (const float*)d_in[3];
  const float* w_conv2  = (const float*)d_in[4];
  const float* bn2_s    = (const float*)d_in[5];
  const float* bn2_b    = (const float*)d_in[6];
  const float* w_mp[4]  = {(const float*)d_in[7], (const float*)d_in[8],
                           (const float*)d_in[9], (const float*)d_in[10]};
  const float* w_seg    = (const float*)d_in[11];
  const float* b_seg    = (const float*)d_in[12];
  const float* w_fc1    = (const float*)d_in[13];
  const float* b_fc1    = (const float*)d_in[14];
  const float* w_fc2    = (const float*)d_in[15];
  const float* b_fc2    = (const float*)d_in[16];
  float* out = (float*)d_out;

  char* ws = (char*)d_ws;
  size_t off = 0;
  auto alloc = [&](size_t bytes) -> void* {
    void* ptr = ws + off;
    off += (bytes + 255) & ~(size_t)255;
    return ptr;
  };

  bf16*  featB  = (bf16*) alloc((size_t)MM*C0*2);
  bf16*  w1T    = (bf16*) alloc((size_t)C1*K1*2);
  bf16*  w2T    = (bf16*) alloc((size_t)C2*C1*2);
  bf16*  wmpT[4];
  for (int i = 0; i < 4; ++i) wmpT[i] = (bf16*)alloc((size_t)C2*KMP*2);
  float* conv1o = (float*)alloc((size_t)MM*C1*4);
  bf16*  act1   = (bf16*) alloc((size_t)MM*C1*2);
  float* x      = (float*)alloc((size_t)MM*C2*4);
  float* stats  = (float*)alloc(2304*4);   // sum1[1024] sq1[1024] sum2[128] sq2[128]
  float* segb   = (float*)alloc((size_t)MM*NSEG*4);
  float* pbuf   = (float*)alloc((size_t)BB*4500*4);
  float* zbuf   = (float*)alloc((size_t)BB*C2*4);

  // --- precision conversions (one-time) ---
  k_cvt_bf16 <<<4096, 256, 0, stream>>>(features, featB, (size_t)MM*C0);
  k_cvtT_bf16<<<4096, 256, 0, stream>>>(w_conv1, w1T, K1, C1);
  k_cvtT_bf16<<<512,  256, 0, stream>>>(w_conv2, w2T, C1, C2);
  for (int i = 0; i < 4; ++i)
    k_cvtT_bf16<<<256, 256, 0, stream>>>(w_mp[i], wmpT[i], KMP, C2);
  k_zero<<<9, 256, 0, stream>>>(stats, 2304);

  // --- conv1 (implicit GEMM, 543 GFLOP -> WMMA bf16) ---
  {
    GArgs g = {};
    g.feat = featB; g.BT = w1T; g.C = conv1o;
    g.M = MM; g.N = C1; g.K = K1;
    k_gemm<1><<<dim3(MM/64, C1/64), 64, 0, stream>>>(g);
  }
  k_bn_reduce<<<MM/64, 256, 0, stream>>>(conv1o, MM, C1, 64, stats, stats+1024);
  k_bn_apply<1><<<8192, 256, 0, stream>>>(conv1o, stats, stats+1024, bn1_s, bn1_b,
                                          1.f/(float)MM, C1, (size_t)MM*C1, nullptr, act1);

  // --- conv2 (1x1, plain GEMM) ---
  {
    GArgs g = {};
    g.A = act1; g.BT = w2T; g.C = x;
    g.M = MM; g.N = C2; g.K = C1;
    k_gemm<0><<<dim3(MM/64, C2/64), 64, 0, stream>>>(g);
  }
  k_bn_reduce<<<MM/64, 256, 0, stream>>>(x, MM, C2, 64, stats+2048, stats+2048+128);
  k_bn_apply<0><<<4096, 256, 0, stream>>>(x, stats+2048, stats+2048+128, bn2_s, bn2_b,
                                          1.f/(float)MM, C2, (size_t)MM*C2, x, nullptr);

  // --- message passing: sequential per-row WMMA GEMM steps (scan via stream order) ---
  auto mp_step = [&](const bf16* wT, float* prev, float* cur, int64_t sP, int L) {
    GArgs g = {};
    g.prev = prev; g.cur = cur;
    g.sB = (int64_t)HH*WW*C2; g.sP = sP; g.L = L;
    g.BT = wT;
    g.M = BB*L; g.N = C2; g.K = KMP;
    k_gemm<2><<<dim3((BB*L)/64, C2/64), 64, 0, stream>>>(g);
  };
  // vertical forward: x[:,i] += relu(conv(x[:,i-1]))
  for (int i = 1; i < HH; ++i)
    mp_step(wmpT[0], x + (size_t)(i-1)*WW*C2, x + (size_t)i*WW*C2, C2, WW);
  // vertical reverse: x[:,i] += relu(conv(x[:,i+1]))
  for (int i = HH-2; i >= 0; --i)
    mp_step(wmpT[1], x + (size_t)(i+1)*WW*C2, x + (size_t)i*WW*C2, C2, WW);
  // horizontal forward: x[:,:,j] += relu(conv(x[:,:,j-1]))
  for (int j = 1; j < WW; ++j)
    mp_step(wmpT[2], x + (size_t)(j-1)*C2, x + (size_t)j*C2, (int64_t)WW*C2, HH);
  // horizontal reverse
  for (int j = WW-2; j >= 0; --j)
    mp_step(wmpT[3], x + (size_t)(j+1)*C2, x + (size_t)j*C2, (int64_t)WW*C2, HH);

  // --- heads ---
  k_seg<<<MM/256, 256, 0, stream>>>(x, w_seg, b_seg, segb);
  k_resize<<<(unsigned)(((size_t)BB*288*800 + 255)/256), 256, 0, stream>>>(segb, out);
  k_softpool<<<(BB*18*50 + 255)/256, 256, 0, stream>>>(segb, pbuf);
  k_fc1<<<BB, C2, 0, stream>>>(pbuf, w_fc1, b_fc1, zbuf);
  k_fc2<<<1, 64, 0, stream>>>(zbuf, w_fc2, b_fc2, out + SEGOUT);
}
